// VideoDiTAttention_67199058313906
// MI455X (gfx1250) — matrix operand order repaired and checked by reading, not compiled, not run on hardware
//
#include <hip/hip_runtime.h>
#include <hip/hip_bf16.h>

// ---------------------------------------------------------------------------
// VideoDiT attention block for MI455X (gfx1250, wave32, WMMA bf16)
//   x[1,4096,1024] -> qkv GEMM -> RoPE -> flash attention -> proj GEMM
// All matmuls via v_wmma_f32_16x16x32_bf16 (f32 accumulate).
// Proj-GEMM A tiles staged via the Tensor Data Mover (tensor_load_to_lds).
// ---------------------------------------------------------------------------

typedef __attribute__((ext_vector_type(16))) __bf16 v16bf;
typedef __attribute__((ext_vector_type(8)))  float  v8f;
typedef __attribute__((ext_vector_type(4)))  unsigned int u32x4;
typedef __attribute__((ext_vector_type(8)))  int i32x8;
typedef __attribute__((ext_vector_type(4)))  int i32x4;

#define SEQ     4096
#define MODEL   1024
#define NHEADS  16
#define HDIM    64
#define QKVDIM  3072
#define ATT_SCALE 0.125f   // 1/sqrt(64)

#if __has_builtin(__builtin_amdgcn_tensor_load_to_lds)
#define HAVE_TDM 1
#else
#define HAVE_TDM 0
#endif

static __device__ __forceinline__ v8f wmma_bf16(v16bf a, v16bf b, v8f c) {
  return __builtin_amdgcn_wmma_f32_16x16x32_bf16(false, a, false, b, (short)0, c,
                                                 false, false);
}

// XOR-butterfly max across lanes via ds_swizzle_b32 (group-of-32 mode:
// and=0x1F, or=0, xor=M; masks 1/2/4/8 never cross the 16-lane C-layout halves)
template <int M>
static __device__ __forceinline__ float xmax(float v) {
  int t = __builtin_amdgcn_ds_swizzle(__float_as_int(v), 0x1F | (M << 10));
  return fmaxf(v, __int_as_float(t));
}

// ===========================================================================
// Tiled GEMM:  Out[M,N] = A[M,K] * B[K,N] + bias[N]
// Block tile 128x128, K-step 64 (two WMMA K-slabs per staging round).
// 8 waves in a 2(M) x 4(N) grid; each wave owns 64x32 = 4x2 accumulators.
// A staged in LDS row-major [128][64] (K contiguous -> A-frag = 2x b128);
//   for bf16 A this staging is done by the TDM (async tensor DMA with LDS
//   padding: 4 DWORDs of pad per 32 DWORDs = our [128][72] layout).
// B staged in LDS transposed [128 n][64 k] (K contiguous -> B-frag = 2x b128).
// ===========================================================================
template <bool A_BF16, bool OUT_BF16>
__global__ __launch_bounds__(256)
void gemm_bias_wmma(const void* __restrict__ Aptr,
                    const float* __restrict__ B,
                    const float* __restrict__ bias,
                    void* __restrict__ Out,
                    int M, int N, int K) {
  __shared__ __bf16 lds_a[128][72];   // 64 + 8 pad: 144B row stride (16B aligned)
  __shared__ __bf16 lds_b[128][72];

  const int tid  = threadIdx.x;
  const int wave = tid >> 5;
  const int lane = tid & 31;
  const int lh   = lane >> 4;
  const int ln   = lane & 15;
  const int wm   = (wave >> 2) * 64;
  const int wn   = (wave & 3) * 32;
  const int m0   = blockIdx.y * 128;
  const int n0   = blockIdx.x * 128;

  v8f acc[4][2];
#pragma unroll
  for (int i = 0; i < 4; ++i)
#pragma unroll
    for (int j = 0; j < 2; ++j)
#pragma unroll
      for (int r = 0; r < 8; ++r) acc[i][j][r] = 0.0f;

  for (int k0 = 0; k0 < K; k0 += 64) {
    // ---- stage A tile (128 x 64) into LDS as bf16 ----
    if constexpr (A_BF16) {
#if HAVE_TDM
      // Tensor Data Mover: one D# issued by wave 0 copies the whole
      // 128x64 bf16 tile (row stride K) into lds_a with LDS padding
      // (pad_interval=32 DWORDs, pad_amount=4 DWORDs -> 144B row stride).
      if (tid < 32) {
        const __bf16* A = (const __bf16*)Aptr;
        unsigned long long ga =
            (unsigned long long)(uintptr_t)&A[(size_t)m0 * K + k0];
        unsigned lds0 = (unsigned)(uintptr_t)&lds_a[0][0];
        u32x4 g0;
        g0[0] = 1u;                                   // count=1 (valid user D#)
        g0[1] = lds0;                                 // lds_addr
        g0[2] = (unsigned)ga;                         // global_addr[31:0]
        g0[3] = (unsigned)((ga >> 32) & 0x01FFFFFFu)  // global_addr[56:32]
              | (2u << 30);                           // type=2 ("image")
        i32x8 g1;
        g1[0] = (1 << 16)      // data_size = 2 bytes
              | (1 << 20)      // pad_enable
              | (4 << 22)      // pad_interval: 32 DWORDs (128B)
              | (3 << 25);     // pad_amount:   4 DWORDs (16B)
        g1[1] = (int)(((unsigned)K & 0xFFFFu) << 16);            // tensor_dim0 lo
        g1[2] = (int)(((unsigned)K >> 16) | ((unsigned)M << 16)); // dim0 hi | dim1 lo
        g1[3] = (int)(((unsigned)M >> 16) | (64u << 16));        // dim1 hi | tile_dim0=64
        g1[4] = 128;                                             // tile_dim1=128
        g1[5] = K;                                               // tensor_dim0_stride lo
        g1[6] = 0;
        g1[7] = 0;
        i32x4 gz = {0, 0, 0, 0};
        i32x8 gz8 = {0, 0, 0, 0, 0, 0, 0, 0};
        __builtin_amdgcn_tensor_load_to_lds(g0, g1, gz, gz, gz8, 0);
        __builtin_amdgcn_s_wait_tensorcnt(0);
      }
#else
      const __bf16* A = (const __bf16*)Aptr;
#pragma unroll
      for (int it = 0; it < 4; ++it) {              // 1024 x (8 bf16) chunks
        int idx = it * 256 + tid;
        int row = idx >> 3;
        int c8  = (idx & 7) * 8;
        uint4 v = *(const uint4*)&A[(size_t)(m0 + row) * K + k0 + c8];
        *(uint4*)&lds_a[row][c8] = v;
      }
#endif
    } else {
      const float* A = (const float*)Aptr;
#pragma unroll
      for (int it = 0; it < 8; ++it) {              // 2048 x (float4) chunks
        int idx = it * 256 + tid;
        int row = idx >> 4;
        int c4  = (idx & 15) * 4;
        float4 v = *(const float4*)&A[(size_t)(m0 + row) * K + k0 + c4];
        union { __bf16 h[4]; uint2 u; } pk;
        pk.h[0] = (__bf16)v.x; pk.h[1] = (__bf16)v.y;
        pk.h[2] = (__bf16)v.z; pk.h[3] = (__bf16)v.w;
        *(uint2*)&lds_a[row][c4] = pk.u;
      }
      if (k0 + 64 < K)                               // global_prefetch_b8
        __builtin_prefetch(&A[(size_t)(m0 + (tid >> 1)) * K + k0 + 64 + (tid & 1) * 32], 0, 3);
    }
    // ---- stage B tile (64 x 128) transposed into LDS: lds_b[n][k] ----
#pragma unroll
    for (int it = 0; it < 8; ++it) {
      int idx = it * 256 + tid;
      int kk  = idx >> 5;
      int nn  = (idx & 31) * 4;
      float4 v = *(const float4*)&B[(size_t)(k0 + kk) * N + n0 + nn];
      lds_b[nn + 0][kk] = (__bf16)v.x;
      lds_b[nn + 1][kk] = (__bf16)v.y;
      lds_b[nn + 2][kk] = (__bf16)v.z;
      lds_b[nn + 3][kk] = (__bf16)v.w;
    }
    if (k0 + 64 < K)
      __builtin_prefetch(&B[(size_t)(k0 + 64 + (tid >> 5)) * N + n0 + (tid & 31) * 4], 0, 3);
    __syncthreads();

    // ---- two WMMA K-slabs per staging round ----
#pragma unroll
    for (int kk2 = 0; kk2 < 64; kk2 += 32) {
      v16bf af[4];
#pragma unroll
      for (int ms = 0; ms < 4; ++ms) {
        int row = wm + ms * 16 + ln;
        ((uint4*)&af[ms])[0] = *(const uint4*)&lds_a[row][kk2 + lh * 8];
        ((uint4*)&af[ms])[1] = *(const uint4*)&lds_a[row][kk2 + 16 + lh * 8];
      }
      v16bf bf[2];
#pragma unroll
      for (int ns = 0; ns < 2; ++ns) {
        int col = wn + ns * 16 + ln;
        ((uint4*)&bf[ns])[0] = *(const uint4*)&lds_b[col][kk2 + lh * 8];
        ((uint4*)&bf[ns])[1] = *(const uint4*)&lds_b[col][kk2 + 16 + lh * 8];
      }
#pragma unroll
      for (int ms = 0; ms < 4; ++ms)
#pragma unroll
        for (int ns = 0; ns < 2; ++ns)
          acc[ms][ns] = wmma_bf16(af[ms], bf[ns], acc[ms][ns]);
    }
    __syncthreads();
  }

  // ---- epilogue: bias, store ----
#pragma unroll
  for (int ns = 0; ns < 2; ++ns) {
    int col = n0 + wn + ns * 16 + ln;
    float bv = bias[col];
#pragma unroll
    for (int ms = 0; ms < 4; ++ms)
#pragma unroll
      for (int r = 0; r < 8; ++r) {
        int row = m0 + wm + ms * 16 + lh * 8 + r;
        float v = acc[ms][ns][r] + bv;
        if constexpr (OUT_BF16)
          ((__bf16*)Out)[(size_t)row * N + col] = (__bf16)v;
        else
          ((float*)Out)[(size_t)row * N + col] = v;
      }
  }
}

// ===========================================================================
// RoPE + re-layout. qkv bf16 [4096][3072] ->
//   Qr,Kr: [h][n][64] (row-major, rotated)   Vt: [h][64][n] (transposed)
// ===========================================================================
__global__ __launch_bounds__(256)
void rope_relayout(const __bf16* __restrict__ qkv,
                   const float* __restrict__ cosp,
                   const float* __restrict__ sinp,
                   __bf16* __restrict__ Qr, __bf16* __restrict__ Kr,
                   __bf16* __restrict__ Vt) {
  int t = blockIdx.x * blockDim.x + threadIdx.x;   // < 4096*16*32
  int i = t & 31;
  int h = (t >> 5) & 15;
  int n = t >> 9;
  float c = cosp[n * 32 + i];
  float s = sinp[n * 32 + i];
  size_t base = (size_t)n * QKVDIM + h * HDIM + i;
  float q1 = (float)qkv[base],        q2 = (float)qkv[base + 32];
  float k1 = (float)qkv[base + 1024], k2 = (float)qkv[base + 1056];
  float v1 = (float)qkv[base + 2048], v2 = (float)qkv[base + 2080];
  size_t qo = ((size_t)(h * SEQ + n)) * HDIM + i;
  Qr[qo]      = (__bf16)(q1 * c - q2 * s);
  Qr[qo + 32] = (__bf16)(q2 * c + q1 * s);
  Kr[qo]      = (__bf16)(k1 * c - k2 * s);
  Kr[qo + 32] = (__bf16)(k2 * c + k1 * s);
  size_t vo = ((size_t)(h * HDIM + i)) * SEQ + n;
  Vt[vo]                    = (__bf16)v1;
  Vt[vo + (size_t)32 * SEQ] = (__bf16)v2;
}

// ===========================================================================
// Flash attention, one wave32 per (head, 16-query tile), 64-key tiles.
//   S = Q K^T (4 key-subtiles x 2 K-steps = 8 WMMA)
//   online softmax: row-max via 4 ds_swizzle XOR steps (amortized over 64
//   keys); row-sum via P @ ones WMMA (2) -- no cross-lane sum reduction.
//   O += P V (4 dim-subtiles x 2 K-steps = 8 WMMA).
// ===========================================================================
__global__ __launch_bounds__(32)
void flash_attn_wmma(const __bf16* __restrict__ Qr,
                     const __bf16* __restrict__ Kr,
                     const __bf16* __restrict__ Vt,
                     __bf16* __restrict__ attnb) {
  __shared__ __bf16 p_lds[16][64];   // 2KB, 128B row stride
  const int lane = threadIdx.x;
  const int lh = lane >> 4, ln = lane & 15;
  const int h  = blockIdx.y;
  const int qb = blockIdx.x * 16;

  // Q fragments (A-matrix 16x32, two K-steps covering head_dim 64)
  v16bf qf[2];
  const __bf16* qrow = Qr + ((size_t)(h * SEQ + qb + ln)) * HDIM;
#pragma unroll
  for (int ks = 0; ks < 2; ++ks) {
    ((uint4*)&qf[ks])[0] = *(const uint4*)(qrow + ks * 32 + lh * 8);
    ((uint4*)&qf[ks])[1] = *(const uint4*)(qrow + ks * 32 + 16 + lh * 8);
  }
  // all-ones B fragment (row-sum via WMMA)
  v16bf ones;
#pragma unroll
  for (int i = 0; i < 16; ++i) ones[i] = (__bf16)1.0f;

  v8f o[4], osum;
  float m_i[8];
#pragma unroll
  for (int dt = 0; dt < 4; ++dt)
#pragma unroll
    for (int r = 0; r < 8; ++r) o[dt][r] = 0.0f;
#pragma unroll
  for (int r = 0; r < 8; ++r) { osum[r] = 0.0f; m_i[r] = -1e30f; }

  for (int kb = 0; kb < SEQ; kb += 64) {
    // ---- S = Q K^T for 64 keys (4 subtiles of 16) ----
    v8f sacc[4];
#pragma unroll
    for (int nt = 0; nt < 4; ++nt) {
#pragma unroll
      for (int r = 0; r < 8; ++r) sacc[nt][r] = 0.0f;
#pragma unroll
      for (int ks = 0; ks < 2; ++ks) {
        v16bf kf;
        const __bf16* krow =
            Kr + ((size_t)(h * SEQ + kb + nt * 16 + ln)) * HDIM + ks * 32 + lh * 8;
        ((uint4*)&kf)[0] = *(const uint4*)(krow);
        ((uint4*)&kf)[1] = *(const uint4*)(krow + 16);
        sacc[nt] = wmma_bf16(qf[ks], kf, sacc[nt]);
      }
    }
    if (kb + 64 < SEQ) {   // prefetch next K tile
      __builtin_prefetch(Kr + ((size_t)(h * SEQ + kb + 64 + lane)) * HDIM, 0, 3);
      __builtin_prefetch(Vt + ((size_t)(h * HDIM + lane)) * SEQ + kb + 64, 0, 3);
    }

    // ---- online softmax over 64 keys ----
#pragma unroll
    for (int r = 0; r < 8; ++r) {
      float s0 = sacc[0][r] * ATT_SCALE;
      float s1 = sacc[1][r] * ATT_SCALE;
      float s2 = sacc[2][r] * ATT_SCALE;
      float s3 = sacc[3][r] * ATT_SCALE;
      float mx = fmaxf(fmaxf(s0, s1), fmaxf(s2, s3));
      mx = xmax<1>(mx); mx = xmax<2>(mx); mx = xmax<4>(mx); mx = xmax<8>(mx);
      float mn = fmaxf(m_i[r], mx);
      float alpha = __expf(m_i[r] - mn);
      m_i[r] = mn;
      float p0 = __expf(s0 - mn);
      float p1 = __expf(s1 - mn);
      float p2 = __expf(s2 - mn);
      float p3 = __expf(s3 - mn);
      osum[r] *= alpha;
#pragma unroll
      for (int dt = 0; dt < 4; ++dt) o[dt][r] *= alpha;
      int row = lh * 8 + r;
      p_lds[row][ln]      = (__bf16)p0;
      p_lds[row][16 + ln] = (__bf16)p1;
      p_lds[row][32 + ln] = (__bf16)p2;
      p_lds[row][48 + ln] = (__bf16)p3;
    }
    __syncthreads();

    // ---- P fragments (A-matrix 16x32 x2) ----
    v16bf pf[2];
#pragma unroll
    for (int ks = 0; ks < 2; ++ks) {
      ((uint4*)&pf[ks])[0] = *(const uint4*)&p_lds[ln][ks * 32 + lh * 8];
      ((uint4*)&pf[ks])[1] = *(const uint4*)&p_lds[ln][ks * 32 + 16 + lh * 8];
    }
    // row-sum of P via ones-column WMMA
    osum = wmma_bf16(pf[0], ones, osum);
    osum = wmma_bf16(pf[1], ones, osum);

    // ---- O += P V ----
#pragma unroll
    for (int dt = 0; dt < 4; ++dt) {
#pragma unroll
      for (int ks = 0; ks < 2; ++ks) {
        v16bf vf;
        const __bf16* vrow =
            Vt + ((size_t)(h * HDIM + dt * 16 + ln)) * SEQ + kb + ks * 32 + lh * 8;
        ((uint4*)&vf)[0] = *(const uint4*)(vrow);
        ((uint4*)&vf)[1] = *(const uint4*)(vrow + 16);
        o[dt] = wmma_bf16(pf[ks], vf, o[dt]);
      }
    }
    __syncthreads();
  }

  // ---- normalize + store bf16 [n][1024] ----
#pragma unroll
  for (int dt = 0; dt < 4; ++dt) {
    int col = h * HDIM + dt * 16 + ln;
#pragma unroll
    for (int r = 0; r < 8; ++r) {
      int row = qb + lh * 8 + r;
      attnb[(size_t)row * MODEL + col] = (__bf16)(o[dt][r] / osum[r]);
    }
  }
}

// ===========================================================================
// Launch. Workspace layout (bytes):
//   qkv  bf16 [4096][3072] : 25,165,824
//   Qr   bf16 [16][4096][64], Kr same, Vt bf16 [16][64][4096] : 3 x 8,388,608
//   attn bf16 [4096][1024] : 8,388,608     (total ~58.7 MB)
// ===========================================================================
extern "C" void kernel_launch(void* const* d_in, const int* in_sizes, int n_in,
                              void* d_out, int out_size, void* d_ws, size_t ws_size,
                              hipStream_t stream) {
  const float* x      = (const float*)d_in[0];
  const float* cosp   = (const float*)d_in[1];
  const float* sinp   = (const float*)d_in[2];
  const float* w_qkv  = (const float*)d_in[3];
  const float* b_qkv  = (const float*)d_in[4];
  const float* w_proj = (const float*)d_in[5];
  const float* b_proj = (const float*)d_in[6];
  float* out = (float*)d_out;

  char* ws = (char*)d_ws;
  __bf16* qkv = (__bf16*)ws;
  __bf16* Qr  = (__bf16*)(ws + (size_t)SEQ * QKVDIM * 2);
  __bf16* Kr  = Qr + (size_t)NHEADS * SEQ * HDIM;
  __bf16* Vt  = Kr + (size_t)NHEADS * SEQ * HDIM;
  __bf16* attnb = Vt + (size_t)NHEADS * SEQ * HDIM;

  gemm_bias_wmma<false, true><<<dim3(QKVDIM / 128, SEQ / 128), 256, 0, stream>>>(
      x, w_qkv, b_qkv, qkv, SEQ, QKVDIM, MODEL);

  rope_relayout<<<(SEQ * NHEADS * 32) / 256, 256, 0, stream>>>(
      qkv, cosp, sinp, Qr, Kr, Vt);

  flash_attn_wmma<<<dim3(SEQ / 16, NHEADS), 32, 0, stream>>>(Qr, Kr, Vt, attnb);

  gemm_bias_wmma<true, false><<<dim3(MODEL / 128, SEQ / 128), 256, 0, stream>>>(
      attnb, w_proj, b_proj, out, SEQ, MODEL, MODEL);
}
